// SpMiddleFHD_82729660055563
// MI455X (gfx1250) — matrix operand order repaired and checked by reading, not compiled, not run on hardware
//
#include <hip/hip_runtime.h>
#include <hip/hip_bf16.h>
#include <hip/hip_fp16.h>

// ---------------------------------------------------------------------------
// CDNA5 (gfx1250) implicit-GEMM sparse-middle-extractor pipeline.
// Dense NDHWC f16 activations, WMMA f32_16x16x32_f16 matrix cores.
// K-extent (taps * cin) is a compile-time constant per layer so the inner
// loop fully unrolls: constant tap offsets, immediate-offset b128 loads.
// ---------------------------------------------------------------------------

typedef __attribute__((ext_vector_type(16))) _Float16 v16h;
typedef __attribute__((ext_vector_type(8)))  float    v8f;
typedef __attribute__((ext_vector_type(4)))  unsigned v4u;

union HFrag { v4u q[2]; v16h h; };

static __device__ __forceinline__ unsigned short f2h_bits(float f) {
    union { _Float16 h; unsigned short u; } cv; cv.h = (_Float16)f; return cv.u;
}

// ------------------------------ scatter ------------------------------------
__global__ void scatter_kernel(const float* __restrict__ feats,
                               const int* __restrict__ coors,
                               _Float16* __restrict__ act,
                               unsigned char* __restrict__ mask,
                               int nvox, int C, int D, int H, int W) {
    int i = blockIdx.x * blockDim.x + threadIdx.x;
    if (i >= nvox * C) return;
    int v = i / C;
    int c = i - v * C;
    int z = coors[v * 4 + 1], y = coors[v * 4 + 2], x = coors[v * 4 + 3];
    if ((unsigned)z >= (unsigned)D || (unsigned)y >= (unsigned)H ||
        (unsigned)x >= (unsigned)W) return;
    size_t vox = ((size_t)z * H + y) * W + x;
    act[vox * C + c] = (_Float16)feats[(size_t)v * C + c];
    if (c == 0) mask[vox] = 1;
}

// ------------------------- weight fragment packing -------------------------
// Packs DHWIO f32 weights into the exact B-fragment layout of
// v_wmma_f32_16x16x32_f16: pw[((nb*nkc + kc)*32 + lane)*8 + v] holds the
// fp16 pair (K=8*(lane>>4)+2v+8*(v>>2), +1) for column n = nb*16 + (lane&15).
__global__ void pack_weights_kernel(const float* __restrict__ w,
                                    unsigned* __restrict__ pw,
                                    int cin, int cout, int ntaps, int nkc) {
    int idx = blockIdx.x * blockDim.x + threadIdx.x;
    int total = (cout >> 4) * nkc * 256;
    if (idx >= total) return;
    int v    = idx & 7;
    int lane = (idx >> 3) & 31;
    int kc   = (idx >> 8) % nkc;
    int nb   = (idx >> 8) / nkc;
    int n    = nb * 16 + (lane & 15);
    int kg   = lane >> 4;
    int K0   = kc * 32 + kg * 8 + 2 * v + ((v >> 2) << 3);
    unsigned r = 0;
#pragma unroll
    for (int e = 0; e < 2; ++e) {
        int K   = K0 + e;
        int tap = K / cin;
        int c   = K - tap * cin;
        float val = (tap < ntaps) ? w[((size_t)tap * cin + c) * cout + n] : 0.f;
        r |= (unsigned)f2h_bits(val) << (16 * e);
    }
    pw[idx] = r;
}

// ----------------------------- mask downsample -----------------------------
__global__ void mask_down_kernel(const unsigned char* __restrict__ mi,
                                 unsigned char* __restrict__ mo,
                                 int Din, int Hin, int Win,
                                 int Do, int Ho, int Wo,
                                 int kd, int kh, int kw,
                                 int sd, int sh, int sw,
                                 int pd, int ph, int pw) {
    int i = blockIdx.x * blockDim.x + threadIdx.x;
    int n = Do * Ho * Wo;
    if (i >= n) return;
    int ow = i % Wo; int t = i / Wo; int oh = t % Ho; int od = t / Ho;
    unsigned char r = 0;
    for (int a = 0; a < kd; ++a) {
        int id = od * sd + a - pd;
        if ((unsigned)id >= (unsigned)Din) continue;
        for (int b = 0; b < kh; ++b) {
            int ih = oh * sh + b - ph;
            if ((unsigned)ih >= (unsigned)Hin) continue;
            for (int c = 0; c < kw; ++c) {
                int iw = ow * sw + c - pw;
                if ((unsigned)iw >= (unsigned)Win) continue;
                r |= mi[((size_t)id * Hin + ih) * Win + iw];
            }
        }
    }
    mo[i] = r ? 1 : 0;
}

// --------------------------- WMMA implicit-GEMM conv -----------------------
// One wave -> 16 output voxels x full cout (NBLK accumulators of 16 cols).
// A fragment gathered straight from NDHWC f16 global (two b128 per K-chunk,
// zero-filled for padding / phantom taps); B fragment streamed from the
// pre-packed weights (L2/WGP$ resident). K-loop fully unrolled: all tap
// indices / channel offsets are compile-time constants.
template <int KHW, int KW, int CINSHIFT, int NBLK, int NTAPS>
__global__ __launch_bounds__(256) void conv_wmma_kernel(
    const _Float16* __restrict__ in, _Float16* __restrict__ out,
    const unsigned* __restrict__ pw,
    const float* __restrict__ gsc, const float* __restrict__ bia,
    const unsigned char* __restrict__ mask,
    int Din, int Hin, int Win, int Ho, int Wo, int nvoxOut,
    int sd, int sh, int sw, int pd, int ph, int pww,
    int ntiles) {
    constexpr int CIN = 1 << CINSHIFT;
    constexpr int NKC = (NTAPS * CIN + 31) / 32;   // K-chunks of 32
    int lane = threadIdx.x & 31;
    int wave = blockIdx.x * (blockDim.x >> 5) + (threadIdx.x >> 5);
    if (wave >= ntiles) return;
    int half = lane >> 4;
    int mrow = lane & 15;
    int vm = wave * 16 + mrow;                    // A-row output voxel
    int ow = vm % Wo; int t1 = vm / Wo; int oh = t1 % Ho; int od = t1 / Ho;
    int bd = od * sd - pd, bh = oh * sh - ph, bw = ow * sw - pww;

    v8f acc[NBLK];
#pragma unroll
    for (int nb = 0; nb < NBLK; ++nb) acc[nb] = (v8f)0.0f;

#pragma unroll
    for (int kc = 0; kc < NKC; ++kc) {
        HFrag a;
#pragma unroll
        for (int r = 0; r < 2; ++r) {
            const int kbase = (kc << 5) + (r << 4);   // compile-time
            int kstart = kbase + (half << 3);
            int tap = kstart >> CINSHIFT;
            int c   = kstart & (CIN - 1);
            v4u q = (v4u)0u;
            if (tap < NTAPS) {
                int td  = tap / KHW;
                int rem = tap - td * KHW;
                int th  = rem / KW;
                int tw  = rem - th * KW;
                int id = bd + td, ih = bh + th, iw = bw + tw;
                if ((unsigned)id < (unsigned)Din &&
                    (unsigned)ih < (unsigned)Hin &&
                    (unsigned)iw < (unsigned)Win) {
                    size_t idx = (((size_t)((id * Hin + ih) * Win + iw)) << CINSHIFT) + c;
                    q = *(const v4u*)(in + idx);
                }
            }
            a.q[r] = q;
        }
#pragma unroll
        for (int nb = 0; nb < NBLK; ++nb) {
            const v4u* pb = (const v4u*)pw + ((size_t)(nb * NKC + kc) * 32 + lane) * 2;
            HFrag b;
            b.q[0] = pb[0];
            b.q[1] = pb[1];
            acc[nb] = __builtin_amdgcn_wmma_f32_16x16x32_f16(
                false, a.h, false, b.h, (short)0, acc[nb], false, false);
        }
    }

    // epilogue: relu(acc*g + b) * mask, store f16 NDHWC
    int n0 = lane & 15;
    float mval[8];
#pragma unroll
    for (int v = 0; v < 8; ++v) {
        int voxel = wave * 16 + (half << 3) + v;
        mval[v] = (voxel < nvoxOut) ? (float)mask[voxel] : 0.f;
    }
#pragma unroll
    for (int nb = 0; nb < NBLK; ++nb) {
        int n = nb * 16 + n0;
        float gg = gsc[n], bb = bia[n];
#pragma unroll
        for (int v = 0; v < 8; ++v) {
            int m = (half << 3) + v;                // C/D layout: M = v + 8*half
            int voxel = wave * 16 + m;
            if (voxel < nvoxOut) {
                float x = acc[nb][v] * gg + bb;
                x = x > 0.f ? x : 0.f;
                x *= mval[v];
                out[(size_t)voxel * (NBLK * 16) + n] = (_Float16)x;
            }
        }
    }
}

// ------------------------------ final permute ------------------------------
// act: [d][h][w][c] f16  ->  out: [c*D + d][h][w] f32
__global__ void final_permute_kernel(const _Float16* __restrict__ act,
                                     float* __restrict__ out,
                                     int D, int H, int W, int C) {
    int i = blockIdx.x * blockDim.x + threadIdx.x;
    int total = D * H * W * C;
    if (i >= total) return;
    int c = i % C; int t = i / C;
    int w = t % W; t /= W;
    int h = t % H; int d = t / H;
    out[(((size_t)c * D + d) * H + h) * W + w] = (float)act[i];
}

// ------------------------------- host driver -------------------------------
extern "C" void kernel_launch(void* const* d_in, const int* in_sizes, int n_in,
                              void* d_out, int out_size, void* d_ws, size_t ws_size,
                              hipStream_t stream) {
    (void)in_sizes; (void)n_in; (void)out_size; (void)ws_size;
    const float* feats = (const float*)d_in[0];
    const int*   coors = (const int*)d_in[1];
    // d_in[2] = batch_size (always 1), ignored.

    struct LD { int down, cin, cout, kd, kh, kw, sd, sh, sw, pD, pH, pW; };
    static const LD L[14] = {
        {0, 16, 16, 3, 3, 3, 1, 1, 1, 1, 1, 1},
        {0, 16, 16, 3, 3, 3, 1, 1, 1, 1, 1, 1},
        {1, 16, 32, 3, 3, 3, 2, 2, 2, 1, 1, 1},
        {0, 32, 32, 3, 3, 3, 1, 1, 1, 1, 1, 1},
        {0, 32, 32, 3, 3, 3, 1, 1, 1, 1, 1, 1},
        {1, 32, 64, 3, 3, 3, 2, 2, 2, 1, 1, 1},
        {0, 64, 64, 3, 3, 3, 1, 1, 1, 1, 1, 1},
        {0, 64, 64, 3, 3, 3, 1, 1, 1, 1, 1, 1},
        {0, 64, 64, 3, 3, 3, 1, 1, 1, 1, 1, 1},
        {1, 64, 64, 3, 3, 3, 2, 2, 2, 1, 1, 1},
        {0, 64, 64, 3, 3, 3, 1, 1, 1, 1, 1, 1},
        {0, 64, 64, 3, 3, 3, 1, 1, 1, 1, 1, 1},
        {0, 64, 64, 3, 3, 3, 1, 1, 1, 1, 1, 1},
        {1, 64, 64, 3, 1, 1, 2, 1, 1, 0, 0, 0},
    };

    // resolutions
    int RD[5], RH[5], RW[5];
    RD[0] = 41; RH[0] = 200; RW[0] = 176;
    {
        int r = 0;
        for (int i = 0; i < 14; ++i) {
            if (L[i].down) {
                RD[r + 1] = (RD[r] + 2 * L[i].pD - L[i].kd) / L[i].sd + 1;
                RH[r + 1] = (RH[r] + 2 * L[i].pH - L[i].kh) / L[i].sh + 1;
                RW[r + 1] = (RW[r] + 2 * L[i].pW - L[i].kw) / L[i].sw + 1;
                ++r;
            }
        }
    }

    // workspace carving
    char* ws = (char*)d_ws;
    size_t off = 0;
    auto alloc = [&](size_t bytes) -> char* {
        char* p = ws + off;
        off += (bytes + 255) & ~(size_t)255;
        return p;
    };
    const int NV = 40000, C0 = 16;
    size_t maxActBytes = (size_t)RD[0] * RH[0] * RW[0] * C0 * 2;  // 46.2 MB
    _Float16* bufA = (_Float16*)alloc(maxActBytes);
    _Float16* bufB = (_Float16*)alloc(maxActBytes);
    unsigned char* maskBuf[5];
    for (int q = 0; q < 5; ++q)
        maskBuf[q] = (unsigned char*)alloc((size_t)RD[q] * RH[q] * RW[q]);
    unsigned* pwBuf[14];
    int nkcL[14];
    for (int i = 0; i < 14; ++i) {
        int ntaps = L[i].kd * L[i].kh * L[i].kw;
        int K = ntaps * L[i].cin;
        nkcL[i] = (K + 31) / 32;
        pwBuf[i] = (unsigned*)alloc((size_t)(L[i].cout / 16) * nkcL[i] * 256 * 4);
    }

    // zero-init res0 activation + mask, scatter voxels
    hipMemsetAsync(bufA, 0, (size_t)RD[0] * RH[0] * RW[0] * C0 * 2, stream);
    hipMemsetAsync(maskBuf[0], 0, (size_t)RD[0] * RH[0] * RW[0], stream);
    {
        int tot = NV * C0;
        scatter_kernel<<<(tot + 255) / 256, 256, 0, stream>>>(
            feats, coors, bufA, maskBuf[0], NV, C0, RD[0], RH[0], RW[0]);
    }

    // pack all weights into WMMA B-fragment layout
    for (int i = 0; i < 14; ++i) {
        const float* w = (const float*)d_in[3 + 3 * i + 0];
        int tot = (L[i].cout / 16) * nkcL[i] * 256;
        pack_weights_kernel<<<(tot + 255) / 256, 256, 0, stream>>>(
            w, pwBuf[i], L[i].cin, L[i].cout, L[i].kd * L[i].kh * L[i].kw, nkcL[i]);
    }

    // layer loop
    _Float16* cur = bufA;
    _Float16* nxt = bufB;
    int r = 0;
    for (int i = 0; i < 14; ++i) {
        const LD& l = L[i];
        int Din = RD[r], Hin = RH[r], Win = RW[r];
        int ro = r + (l.down ? 1 : 0);
        int Do = RD[ro], Ho = RH[ro], Wo = RW[ro];
        if (l.down) {
            int n = Do * Ho * Wo;
            mask_down_kernel<<<(n + 255) / 256, 256, 0, stream>>>(
                maskBuf[r], maskBuf[ro], Din, Hin, Win, Do, Ho, Wo,
                l.kd, l.kh, l.kw, l.sd, l.sh, l.sw, l.pD, l.pH, l.pW);
        }
        const float* g = (const float*)d_in[3 + 3 * i + 1];
        const float* b = (const float*)d_in[3 + 3 * i + 2];
        int nvoxOut = Do * Ho * Wo;
        int ntiles = (nvoxOut + 15) / 16;
        int blocks = (ntiles + 7) / 8;   // 8 waves / 256-thread block

#define LAUNCH_CONV(KHW, KW, SH, NB, NT)                                      \
    conv_wmma_kernel<KHW, KW, SH, NB, NT><<<blocks, 256, 0, stream>>>(        \
        cur, nxt, pwBuf[i], g, b, maskBuf[ro], Din, Hin, Win, Ho, Wo,         \
        nvoxOut, l.sd, l.sh, l.sw, l.pD, l.pH, l.pW, ntiles)

        if (l.kh == 3) {
            if (l.cin == 16 && l.cout == 16)       LAUNCH_CONV(9, 3, 4, 1, 27);
            else if (l.cin == 16 && l.cout == 32)  LAUNCH_CONV(9, 3, 4, 2, 27);
            else if (l.cin == 32 && l.cout == 32)  LAUNCH_CONV(9, 3, 5, 2, 27);
            else if (l.cin == 32 && l.cout == 64)  LAUNCH_CONV(9, 3, 5, 4, 27);
            else                                   LAUNCH_CONV(9, 3, 6, 4, 27);
        } else {
            LAUNCH_CONV(1, 1, 6, 4, 3);
        }
#undef LAUNCH_CONV

        _Float16* tmp = cur; cur = nxt; nxt = tmp;
        r = ro;
    }

    // final permute: f16 NDHWC -> f32 (C*D, H, W)
    {
        int tot = RD[4] * RH[4] * RW[4] * 64;
        final_permute_kernel<<<(tot + 255) / 256, 256, 0, stream>>>(
            cur, (float*)d_out, RD[4], RH[4], RW[4], 64);
    }
}
